// ContextualLoss_377957122730
// MI455X (gfx1250) — matrix operand order repaired and checked by reading, not compiled, not run on hardware
//
#include <hip/hip_runtime.h>
#include <hip/hip_bf16.h>
#include <math.h>

typedef __bf16 bf16;
typedef bf16 bf16x8  __attribute__((ext_vector_type(8)));
typedef bf16 bf16x16 __attribute__((ext_vector_type(16)));
typedef float f32x8  __attribute__((ext_vector_type(8)));

#define N_  8
#define C_  512
#define P_  2304        /* 48*48 */
#define EPS_      1e-5f
#define NORM_EPS_ 1e-12f

/* ---------------- kernel 1: channel mean of target over (N,H,W) ------------- */
__global__ void k_mu(const float* __restrict__ tgt, float* __restrict__ mu) {
  int c = blockIdx.x, t = threadIdx.x;
  const int NP = N_ * P_;
  float s = 0.f;
  for (int i = t; i < NP; i += 256) {
    int n = i / P_, p = i - n * P_;
    s += tgt[((size_t)n * C_ + c) * P_ + p];
  }
  __shared__ float red[256];
  red[t] = s; __syncthreads();
  for (int off = 128; off > 0; off >>= 1) {
    if (t < off) red[t] += red[t + off];
    __syncthreads();
  }
  if (t == 0) mu[c] = red[0] / (float)NP;
}

/* ------------- kernel 2: per-(n,p) inverse L2 norms along C ----------------- */
__global__ void k_norms(const float* __restrict__ pred, const float* __restrict__ tgt,
                        const float* __restrict__ mu,
                        float* __restrict__ invx, float* __restrict__ invy) {
  int n = blockIdx.y;
  int p = blockIdx.x * 256 + threadIdx.x;
  float sx = 0.f, sy = 0.f;
  for (int c = 0; c < C_; ++c) {
    float m = mu[c];
    float a = pred[((size_t)n * C_ + c) * P_ + p] - m;
    float b = tgt [((size_t)n * C_ + c) * P_ + p] - m;
    sx += a * a; sy += b * b;
  }
  invx[n * P_ + p] = 1.f / fmaxf(sqrtf(sx), NORM_EPS_);
  invy[n * P_ + p] = 1.f / fmaxf(sqrtf(sy), NORM_EPS_);
}

/* ------- kernel 3: center, normalize, transpose [C,P]->[P,C], cvt bf16 ------ */
__global__ void k_pack(const float* __restrict__ pred, const float* __restrict__ tgt,
                       const float* __restrict__ mu,
                       const float* __restrict__ invx, const float* __restrict__ invy,
                       bf16* __restrict__ Xt, bf16* __restrict__ Yt) {
  __shared__ bf16 tx[32][34];
  __shared__ bf16 ty[32][34];
  int n = blockIdx.z;
  int c0 = blockIdx.y * 32;
  int p0 = blockIdx.x * 32;
  int lx = threadIdx.x;                 /* 0..31 */
  for (int r = 0; r < 4; ++r) {
    int cr = threadIdx.y + r * 8;
    int c = c0 + cr;
    int p = p0 + lx;
    float m  = mu[c];
    float ix = invx[n * P_ + p], iy = invy[n * P_ + p];
    tx[cr][lx] = (bf16)((pred[((size_t)n * C_ + c) * P_ + p] - m) * ix);
    ty[cr][lx] = (bf16)((tgt [((size_t)n * C_ + c) * P_ + p] - m) * iy);
  }
  __syncthreads();
  for (int r = 0; r < 4; ++r) {
    int pr = threadIdx.y + r * 8;
    size_t o = ((size_t)n * P_ + (p0 + pr)) * C_ + (c0 + lx);
    Xt[o] = tx[lx][pr];
    Yt[o] = ty[lx][pr];
  }
}

/* ---------------- kernel 4: WMMA GEMM  S = Xt * Yt^T  (per batch) ----------- */
/* Xt: [P,C] row-major bf16 (rows = queries p = M),  Yt: [P,C] (rows = keys q = N).
 * Double-buffered LDS staging with CDNA5 async global->LDS copies (ASYNCcnt).  */
#define LDA_ 40
#define LDB_ 48

__device__ __forceinline__ unsigned lds_off(const void* p) {
  /* ISA 10.2: generic addr of LDS has the wave-relative LDS byte address in [31:0] */
  return (unsigned)(uintptr_t)p;
}
__device__ __forceinline__ void async_ld16(unsigned lds, const bf16* g) {
  asm volatile("global_load_async_to_lds_b128 %0, %1, off"
               :: "v"(lds), "v"((unsigned long long)(uintptr_t)g) : "memory");
}

__global__ void __launch_bounds__(256)
k_gemm(const bf16* __restrict__ Xt, const bf16* __restrict__ Yt,
       float* __restrict__ S) {
  __shared__ __align__(16) bf16 lA[2][128 * LDA_];
  __shared__ __align__(16) bf16 lB[2][128 * LDB_];
  int n   = blockIdx.z;
  int pm0 = blockIdx.y * 128;
  int pn0 = blockIdx.x * 128;
  const bf16* A = Xt + (size_t)n * P_ * C_;
  const bf16* B = Yt + (size_t)n * P_ * C_;
  int t    = threadIdx.x;
  int lane = t & 31;
  int w    = t >> 5;
  int wm   = (w & 1) * 64;     /* 2 waves along M */
  int wn   = (w >> 1) * 32;    /* 4 waves along N */
  int half = lane >> 4;
  int l16  = lane & 15;

  /* this thread's two 16B chunks per matrix per K-tile (rows r and r+64) */
  int row = t >> 2;
  int off = (t & 3) * 8;
  const bf16* gA0 = A + (size_t)(pm0 + row) * C_ + off;
  const bf16* gA1 = gA0 + (size_t)64 * C_;
  const bf16* gB0 = B + (size_t)(pn0 + row) * C_ + off;
  const bf16* gB1 = gB0 + (size_t)64 * C_;
  unsigned laA[2], laA2[2], laB[2], laB2[2];
  for (int b = 0; b < 2; ++b) {
    laA [b] = lds_off(&lA[b][row * LDA_ + off]);
    laA2[b] = lds_off(&lA[b][(row + 64) * LDA_ + off]);
    laB [b] = lds_off(&lB[b][row * LDB_ + off]);
    laB2[b] = lds_off(&lB[b][(row + 64) * LDB_ + off]);
  }
  auto issue = [&](int buf, int kt) {
    async_ld16(laA [buf], gA0 + kt);
    async_ld16(laA2[buf], gA1 + kt);
    async_ld16(laB [buf], gB0 + kt);
    async_ld16(laB2[buf], gB1 + kt);
  };

  f32x8 acc[4][2];
  f32x8 zero = {0.f,0.f,0.f,0.f,0.f,0.f,0.f,0.f};
  for (int i = 0; i < 4; ++i)
    for (int j = 0; j < 2; ++j) acc[i][j] = zero;

  const int NK = C_ / 32;            /* 16 K-tiles */
  issue(0, 0);
  for (int step = 0; step < NK; ++step) {
    int cur = step & 1;
    if (step + 1 < NK) {
      issue(cur ^ 1, (step + 1) * 32);              /* prefetch next tile */
      asm volatile("s_wait_asynccnt 4" ::: "memory"); /* current tile landed */
    } else {
      asm volatile("s_wait_asynccnt 0" ::: "memory");
    }
    __syncthreads();

    /* B fragments: lane holds N = l16, K blocks [half*8..+7] and [16+half*8..+7] */
    union { bf16x16 v; bf16x8 h[2]; } bfr[2];
    for (int j = 0; j < 2; ++j) {
      const bf16* bp = &lB[cur][(wn + j * 16 + l16) * LDB_ + half * 8];
      bfr[j].h[0] = *(const bf16x8*)(bp);
      bfr[j].h[1] = *(const bf16x8*)(bp + 16);
    }
    for (int i = 0; i < 4; ++i) {
      /* A fragment: lane holds M = l16, K blocks [half*8..+7] and [16+half*8..+7] */
      union { bf16x16 v; bf16x8 h[2]; } afr;
      const bf16* ap = &lA[cur][(wm + i * 16 + l16) * LDA_];
      afr.h[0] = *(const bf16x8*)(ap + half * 8);
      afr.h[1] = *(const bf16x8*)(ap + 16 + half * 8);
      for (int j = 0; j < 2; ++j) {
        acc[i][j] = __builtin_amdgcn_wmma_f32_16x16x32_bf16(
            false, afr.v, false, bfr[j].v, (short)0, acc[i][j], false, false);
      }
    }
    __syncthreads();   /* done reading buf[cur] before step+1 overwrites it */
  }

  /* C/D layout: VGPR r, lanes 0-15 -> M=r, lanes 16-31 -> M=r+8; N=l16 */
  float* So = S + (size_t)n * P_ * P_;
  for (int i = 0; i < 4; ++i) {
    int mrow = pm0 + wm + i * 16 + half * 8;
    for (int j = 0; j < 2; ++j) {
      int ncol = pn0 + wn + j * 16 + l16;
      for (int r = 0; r < 8; ++r)
        So[(size_t)(mrow + r) * P_ + ncol] = acc[i][j][r];
    }
  }
}

/* ------- kernel 5: per-row max + softmax denominator (row cached in LDS) ---- */
__global__ void k_rowstats(const float* __restrict__ S,
                           float* __restrict__ Aarr, float* __restrict__ Rarr) {
  int p = blockIdx.x, n = blockIdx.y, t = threadIdx.x;
  const float* row = S + ((size_t)n * P_ + p) * P_;
  __shared__ float cache[P_];
  __shared__ float red[256];
  float mx = -1e30f;
  for (int q = t; q < P_; q += 256) {
    float v = row[q];
    cache[q] = v;
    mx = fmaxf(mx, v);
  }
  red[t] = mx; __syncthreads();
  for (int off = 128; off > 0; off >>= 1) {
    if (t < off) red[t] = fmaxf(red[t], red[t + off]);
    __syncthreads();
  }
  float m = red[0];
  __syncthreads();
  /* dist_min = 1-m ; a = 2/(dist_min+eps) ; w = exp(2 - a*(1-S)) (max exp ~ 0) */
  float a = 2.f / ((1.f - m) + EPS_);
  float s = 0.f;
  for (int q = t; q < P_; q += 256)
    s += __expf(2.f - a * (1.f - cache[q]));
  red[t] = s; __syncthreads();
  for (int off = 128; off > 0; off >>= 1) {
    if (t < off) red[t] += red[t + off];
    __syncthreads();
  }
  if (t == 0) {
    Aarr[n * P_ + p] = a;
    Rarr[n * P_ + p] = 1.f / red[0];
  }
}

/* ----------------- kernel 6: column max of cx + per-batch sum --------------- */
__global__ void k_init(float* __restrict__ acc) {
  if (threadIdx.x < N_) acc[threadIdx.x] = 0.f;
}

__global__ void k_colmax(const float* __restrict__ S,
                         const float* __restrict__ Aarr, const float* __restrict__ Rarr,
                         float* __restrict__ acc) {
  int n = blockIdx.y;
  int q = blockIdx.x * 256 + threadIdx.x;
  const float* Sn = S + (size_t)n * P_ * P_;
  float cmax = 0.f;
  for (int p = 0; p < P_; ++p) {
    float a = Aarr[n * P_ + p];
    float r = Rarr[n * P_ + p];
    float cx = __expf(2.f - a * (1.f - Sn[(size_t)p * P_ + q])) * r;
    cmax = fmaxf(cmax, cx);
  }
  __shared__ float red[256];
  red[threadIdx.x] = cmax; __syncthreads();
  for (int off = 128; off > 0; off >>= 1) {
    if (threadIdx.x < off) red[threadIdx.x] += red[threadIdx.x + off];
    __syncthreads();
  }
  if (threadIdx.x == 0) atomicAdd(&acc[n], red[0]);
}

/* ------------------------- kernel 7: final scalar loss ---------------------- */
__global__ void k_final(const float* __restrict__ acc, float* __restrict__ out) {
  float s = 0.f;
  for (int n = 0; n < N_; ++n)
    s += -logf(acc[n] / (float)P_ + EPS_);
  out[0] = s / (float)N_;
}

/* ---------------------------------------------------------------------------- */
extern "C" void kernel_launch(void* const* d_in, const int* in_sizes, int n_in,
                              void* d_out, int out_size, void* d_ws, size_t ws_size,
                              hipStream_t stream) {
  const float* pred = (const float*)d_in[0];
  const float* tgt  = (const float*)d_in[1];
  float* out = (float*)d_out;

  char* base = (char*)d_ws;
  size_t off = 0;
  auto alloc = [&](size_t bytes) -> void* {
    void* p = base + off;
    off = (off + bytes + 255) & ~(size_t)255;
    return p;
  };
  float* mu    = (float*)alloc(C_ * sizeof(float));
  float* invx  = (float*)alloc((size_t)N_ * P_ * sizeof(float));
  float* invy  = (float*)alloc((size_t)N_ * P_ * sizeof(float));
  float* Aarr  = (float*)alloc((size_t)N_ * P_ * sizeof(float));
  float* Rarr  = (float*)alloc((size_t)N_ * P_ * sizeof(float));
  float* accv  = (float*)alloc(N_ * sizeof(float));
  bf16*  Xt    = (bf16*) alloc((size_t)N_ * P_ * C_ * sizeof(bf16));
  bf16*  Yt    = (bf16*) alloc((size_t)N_ * P_ * C_ * sizeof(bf16));
  float* S     = (float*)alloc((size_t)N_ * P_ * P_ * sizeof(float));

  k_mu     <<<C_, 256, 0, stream>>>(tgt, mu);
  k_norms  <<<dim3(P_ / 256, N_), 256, 0, stream>>>(pred, tgt, mu, invx, invy);
  k_pack   <<<dim3(P_ / 32, C_ / 32, N_), dim3(32, 8), 0, stream>>>(
               pred, tgt, mu, invx, invy, Xt, Yt);
  k_gemm   <<<dim3(P_ / 128, P_ / 128, N_), 256, 0, stream>>>(Xt, Yt, S);
  k_rowstats<<<dim3(P_, N_), 256, 0, stream>>>(S, Aarr, Rarr);
  k_init   <<<1, 32, 0, stream>>>(accv);
  k_colmax <<<dim3(P_ / 256, N_), 256, 0, stream>>>(S, Aarr, Rarr, accv);
  k_final  <<<1, 1, 0, stream>>>(accv, out);
}
